// Denoiser_25512105738784
// MI455X (gfx1250) — matrix operand order repaired and checked by reading, not compiled, not run on hardware
//
#include <hip/hip_runtime.h>
#include <hip/hip_bf16.h>
#include <math.h>

// ---------------------------------------------------------------------------
// MI455X / gfx1250 implementation. wave32, WMMA 16x16x32 f16 -> f32.
// GEMM A-side activations are kept as padded f16 in HBM and DMA'd into LDS by
// the Tensor Data Mover (tensor_load_to_lds + s_wait_tensorcnt); weight tiles
// are staged branch-free with f32->f16 convert. LDS tiles are stored in
// WMMA-operand-friendly layouts so each fragment is two ds_load_b128.
// ---------------------------------------------------------------------------

typedef __attribute__((ext_vector_type(16))) _Float16 v16h;
typedef __attribute__((ext_vector_type(8)))  _Float16 v8h;
typedef __attribute__((ext_vector_type(8)))  float    v8f;
typedef __attribute__((ext_vector_type(4)))  unsigned v4u;
typedef __attribute__((ext_vector_type(4)))  int      v4i;
typedef __attribute__((ext_vector_type(8)))  int      v8i;

#define WB   2
#define WH   256
#define WW   256
#define WC   31                 // DIM
#define NPB  (WH * WW)          // 65536 tokens per batch
#define NTOK (WB * NPB)         // 131072 tokens
#define EHID 82                 // int(31*2.66)
#define ECH  164                // 2*EHID
#define LDQ  32                 // padded f16 row for 31-channel tensors
#define LDH  84                 // padded f16 row for 82-channel tensors

#if __has_builtin(__builtin_amdgcn_tensor_load_to_lds)
#define HAVE_TDM 1
#else
#define HAVE_TDM 0
#endif

__device__ __forceinline__ float gelu_exact(float x) {
  return 0.5f * x * (1.0f + erff(x * 0.70710678118654752f));
}

__device__ __forceinline__ v16h cat8(v8h lo, v8h hi) {
  return __builtin_shufflevector(lo, hi, 0, 1, 2, 3, 4, 5, 6, 7,
                                 8, 9, 10, 11, 12, 13, 14, 15);
}

#if HAVE_TDM
// 2D f16 tile DMA: rows_rem x cols window (tile_rows x 32) into contiguous
// LDS [tile_rows][32]; hardware zero-fills OOB rows/cols. D# packed per
// CDNA5 ISA (group0: count=1,type=2; group1: dims/strides).
__device__ __forceinline__ void tdm_load_2d_f16(unsigned lds_addr,
                                                const _Float16* gsrc,
                                                int rows_rem, int cols_rem,
                                                int row_stride, int tile_rows)
{
  unsigned long long ga = (unsigned long long)(size_t)gsrc;
  v4u g0;
  g0[0] = 1u;                                        // count=1, user mode
  g0[1] = lds_addr;                                  // lds_addr [63:32]
  g0[2] = (unsigned)(ga & 0xffffffffu);              // global_addr [95:64]
  g0[3] = (unsigned)((ga >> 32) & 0x1ffffffu) | (2u << 30);  // [120:96] | type=2
  v8i g1;
  g1[0] = (int)(1u << 16);                           // data_size=2B
  g1[1] = (int)(((unsigned)cols_rem & 0xffffu) << 16);        // tensor_dim0 lo
  g1[2] = (int)((((unsigned)cols_rem >> 16) & 0xffffu) |
                (((unsigned)rows_rem & 0xffffu) << 16));      // dim0 hi | dim1 lo
  g1[3] = (int)((((unsigned)rows_rem >> 16) & 0xffffu) |
                (32u << 16));                                 // dim1 hi | tile_dim0=32
  g1[4] = (int)((unsigned)tile_rows & 0xffffu);               // tile_dim1
  g1[5] = (int)(unsigned)row_stride;                          // tensor_dim0_stride lo
  g1[6] = 0;
  g1[7] = 0;
  v4i gz = {0, 0, 0, 0};
#if __clang_major__ >= 23
  v8i gz8 = {0, 0, 0, 0, 0, 0, 0, 0};
  __builtin_amdgcn_tensor_load_to_lds(g0, g1, gz, gz, gz8, 0);
#else
  __builtin_amdgcn_tensor_load_to_lds(g0, g1, gz, gz, 0);
#endif
}
#endif

// ---------------------------------------------------------------------------
// shared GEMM pieces: W tile staging (transposed, branch-free) + fragments
// ---------------------------------------------------------------------------
#define TM  128
#define TN  32
#define TKD 32
#define LDT 40   // padded row stride (halves) for ds-staged tiles

__device__ __forceinline__ void stage_w_tile(_Float16 (*sBt)[LDT],
                                             const float* __restrict__ W,
                                             int tid, int n0, int k0,
                                             int N, int K, int ldw, int wTrans)
{
  if (!wTrans) {
#pragma unroll
    for (int i = tid; i < TKD * TN; i += 128) {
      int r = i >> 5, c = i & 31;                    // r=k, c=n (coalesced in n)
      int gk = k0 + r, gn = n0 + c;
      int gkc = gk < K ? gk : K - 1;
      int gnc = gn < N ? gn : N - 1;
      float v = W[(size_t)gkc * ldw + gnc];
      sBt[c][r] = (_Float16)((gk < K && gn < N) ? v : 0.0f);
    }
  } else {
#pragma unroll
    for (int i = tid; i < TKD * TN; i += 128) {
      int r = i >> 5, c = i & 31;                    // r=n, c=k (coalesced in k)
      int gn = n0 + r, gk = k0 + c;
      int gkc = gk < K ? gk : K - 1;
      int gnc = gn < N ? gn : N - 1;
      float v = W[(size_t)gnc * ldw + gkc];
      sBt[r][c] = (_Float16)((gk < K && gn < N) ? v : 0.0f);
    }
  }
}

// ---------------------------------------------------------------------------
// WMMA GEMM, A = padded f16 [M][lda], TDM-staged:  C = A @ W + bias + resid
// ---------------------------------------------------------------------------
__global__ __launch_bounds__(128)
void k_gemm_wmma_tdm(const _Float16* __restrict__ A16, int lda,
                     const float* __restrict__ W,
                     const float* __restrict__ bias, const float* __restrict__ resid,
                     float* __restrict__ C,
                     int M, int N, int K, int ldw, int wTrans)
{
  __shared__ __align__(64) _Float16 sA[TM * 32];     // TDM target, row stride 32
  __shared__ __align__(16) _Float16 sBt[TN][LDT];

  const int tid  = threadIdx.x;
  const int lane = tid & 31;
  const int wid  = tid >> 5;
  const int m0 = blockIdx.x * TM;
  const int n0 = blockIdx.y * TN;
  const int wm = wid * 32;
  const int half = (lane >> 4) & 1;
  const int rl   = lane & 15;

  v8f acc[2][2];
#pragma unroll
  for (int mi = 0; mi < 2; ++mi)
#pragma unroll
    for (int ni = 0; ni < 2; ++ni)
#pragma unroll
      for (int v = 0; v < 8; ++v) acc[mi][ni][v] = 0.0f;

  for (int k0 = 0; k0 < K; k0 += TKD) {
#if HAVE_TDM
    if (wid == 0) {
      // DMA the 128x32 A tile (OOB rows/cols zero-filled by TDM)
      tdm_load_2d_f16((unsigned)(size_t)&sA[0],
                      A16 + (size_t)m0 * lda + k0,
                      M - m0, lda - k0, lda, TM);
#if __has_builtin(__builtin_amdgcn_s_wait_tensorcnt)
      __builtin_amdgcn_s_wait_tensorcnt(0);
#endif
    }
#else
    // cooperative fallback staging of the f16 A tile
#pragma unroll
    for (int i = tid; i < TM * 32; i += 128) {
      int r = i >> 5, c = i & 31;
      int gm = m0 + r, gk = k0 + c;
      int gmc = gm < M ? gm : M - 1;
      int gkc = gk < lda ? gk : lda - 1;
      _Float16 v = A16[(size_t)gmc * lda + gkc];
      sA[r * 32 + c] = (gm < M && gk < lda) ? v : (_Float16)0.0f;
    }
#endif
    stage_w_tile(sBt, W, tid, n0, k0, N, K, ldw, wTrans);
    __syncthreads();

    v16h af[2], bf[2];
#pragma unroll
    for (int mi = 0; mi < 2; ++mi) {
      int row = wm + mi * 16 + rl;
      v8h lo = *(const v8h*)&sA[row * 32 + half * 8];
      v8h hi = *(const v8h*)&sA[row * 32 + 16 + half * 8];
      af[mi] = cat8(lo, hi);
    }
#pragma unroll
    for (int ni = 0; ni < 2; ++ni) {
      int col = ni * 16 + rl;
      v8h lo = *(const v8h*)&sBt[col][half * 8];
      v8h hi = *(const v8h*)&sBt[col][16 + half * 8];
      bf[ni] = cat8(lo, hi);
    }
#pragma unroll
    for (int mi = 0; mi < 2; ++mi)
#pragma unroll
      for (int ni = 0; ni < 2; ++ni)
        acc[mi][ni] = __builtin_amdgcn_wmma_f32_16x16x32_f16(
            false, af[mi], false, bf[ni], (short)0, acc[mi][ni], false, false);
    __syncthreads();
  }

#pragma unroll
  for (int mi = 0; mi < 2; ++mi)
#pragma unroll
    for (int ni = 0; ni < 2; ++ni)
#pragma unroll
      for (int v = 0; v < 8; ++v) {
        int m = m0 + wm + mi * 16 + half * 8 + v;
        int n = n0 + ni * 16 + rl;
        if (m < M && n < N) {
          float x = acc[mi][ni][v];
          if (bias)  x += bias[n];
          if (resid) x += resid[(size_t)m * N + n];
          C[(size_t)m * N + n] = x;
        }
      }
}

// ---------------------------------------------------------------------------
// WMMA GEMM, A = f32 [M][K] staged with convert:  C = A @ W + bias + resid
// ---------------------------------------------------------------------------
__global__ __launch_bounds__(128)
void k_gemm_wmma(const float* __restrict__ A, const float* __restrict__ W,
                 const float* __restrict__ bias, const float* __restrict__ resid,
                 float* __restrict__ C,
                 int M, int N, int K, int ldw, int wTrans)
{
  __shared__ __align__(16) _Float16 sA[TM][LDT];
  __shared__ __align__(16) _Float16 sBt[TN][LDT];

  const int tid  = threadIdx.x;
  const int lane = tid & 31;
  const int wid  = tid >> 5;
  const int m0 = blockIdx.x * TM;
  const int n0 = blockIdx.y * TN;
  const int wm = wid * 32;
  const int half = (lane >> 4) & 1;
  const int rl   = lane & 15;

  v8f acc[2][2];
#pragma unroll
  for (int mi = 0; mi < 2; ++mi)
#pragma unroll
    for (int ni = 0; ni < 2; ++ni)
#pragma unroll
      for (int v = 0; v < 8; ++v) acc[mi][ni][v] = 0.0f;

  for (int k0 = 0; k0 < K; k0 += TKD) {
#pragma unroll
    for (int i = tid; i < TM * TKD; i += 128) {
      int r = i >> 5, c = i & 31;
      int gm = m0 + r, gk = k0 + c;
      int gmc = gm < M ? gm : M - 1;
      int gkc = gk < K ? gk : K - 1;
      float v = A[(size_t)gmc * K + gkc];
      sA[r][c] = (_Float16)((gm < M && gk < K) ? v : 0.0f);
    }
    stage_w_tile(sBt, W, tid, n0, k0, N, K, ldw, wTrans);
    if (k0 + TKD < K) {
      int gm = m0 + tid;
      if (gm < M) __builtin_prefetch(&A[(size_t)gm * K + k0 + TKD], 0, 1);
    }
    __syncthreads();

    v16h af[2], bf[2];
#pragma unroll
    for (int mi = 0; mi < 2; ++mi) {
      int row = wm + mi * 16 + rl;
      v8h lo = *(const v8h*)&sA[row][half * 8];
      v8h hi = *(const v8h*)&sA[row][16 + half * 8];
      af[mi] = cat8(lo, hi);
    }
#pragma unroll
    for (int ni = 0; ni < 2; ++ni) {
      int col = ni * 16 + rl;
      v8h lo = *(const v8h*)&sBt[col][half * 8];
      v8h hi = *(const v8h*)&sBt[col][16 + half * 8];
      bf[ni] = cat8(lo, hi);
    }
#pragma unroll
    for (int mi = 0; mi < 2; ++mi)
#pragma unroll
      for (int ni = 0; ni < 2; ++ni)
        acc[mi][ni] = __builtin_amdgcn_wmma_f32_16x16x32_f16(
            false, af[mi], false, bf[ni], (short)0, acc[mi][ni], false, false);
    __syncthreads();
  }

#pragma unroll
  for (int mi = 0; mi < 2; ++mi)
#pragma unroll
    for (int ni = 0; ni < 2; ++ni)
#pragma unroll
      for (int v = 0; v < 8; ++v) {
        int m = m0 + wm + mi * 16 + half * 8 + v;
        int n = n0 + ni * 16 + rl;
        if (m < M && n < N) {
          float x = acc[mi][ni][v];
          if (bias)  x += bias[n];
          if (resid) x += resid[(size_t)m * N + n];
          C[(size_t)m * N + n] = x;
        }
      }
}

// ---------------------------------------------------------------------------
// Channel-attention Gram: S[b][i][j] = sum_t Kf[b,t,i] * Qf[b,t,j]
// ---------------------------------------------------------------------------
__global__ __launch_bounds__(256)
void k_gram_wmma(const float* __restrict__ Kf, const float* __restrict__ Qf,
                 float* __restrict__ S, int ntok)
{
  __shared__ __align__(16) _Float16 sKt[8][32][LDT];   // [wave][chan][token]
  __shared__ __align__(16) _Float16 sQt[8][32][LDT];
  __shared__ float part[8][32][32];

  const int b = blockIdx.x;
  const float* kp = Kf + (size_t)b * ntok * WC;
  const float* qp = Qf + (size_t)b * ntok * WC;

  const int tid  = threadIdx.x;
  const int lane = tid & 31;
  const int wid  = tid >> 5;
  const int half = (lane >> 4) & 1;
  const int rl   = lane & 15;

  v8f acc[2][2];
#pragma unroll
  for (int mi = 0; mi < 2; ++mi)
#pragma unroll
    for (int ni = 0; ni < 2; ++ni)
#pragma unroll
      for (int v = 0; v < 8; ++v) acc[mi][ni][v] = 0.0f;

  // ntok (65536) is a multiple of the 256-token stride: no tail guards needed.
  for (int t0 = wid * 32; t0 < ntok; t0 += 256) {
    const int t = t0 + lane;
#pragma unroll
    for (int c = 0; c < 32; ++c) {
      float kv = (c < WC) ? kp[(size_t)t * WC + c] : 0.0f;
      float qv = (c < WC) ? qp[(size_t)t * WC + c] : 0.0f;
      sKt[wid][c][lane] = (_Float16)kv;
      sQt[wid][c][lane] = (_Float16)qv;
    }
    v16h af[2], bf[2];
#pragma unroll
    for (int mi = 0; mi < 2; ++mi) {
      int row = mi * 16 + rl;  // channel i
      v8h lo = *(const v8h*)&sKt[wid][row][half * 8];
      v8h hi = *(const v8h*)&sKt[wid][row][16 + half * 8];
      af[mi] = cat8(lo, hi);
    }
#pragma unroll
    for (int ni = 0; ni < 2; ++ni) {
      int col = ni * 16 + rl;  // channel j
      v8h lo = *(const v8h*)&sQt[wid][col][half * 8];
      v8h hi = *(const v8h*)&sQt[wid][col][16 + half * 8];
      bf[ni] = cat8(lo, hi);
    }
#pragma unroll
    for (int mi = 0; mi < 2; ++mi)
#pragma unroll
      for (int ni = 0; ni < 2; ++ni)
        acc[mi][ni] = __builtin_amdgcn_wmma_f32_16x16x32_f16(
            false, af[mi], false, bf[ni], (short)0, acc[mi][ni], false, false);
  }

#pragma unroll
  for (int mi = 0; mi < 2; ++mi)
#pragma unroll
    for (int ni = 0; ni < 2; ++ni)
#pragma unroll
      for (int v = 0; v < 8; ++v)
        part[wid][mi * 16 + half * 8 + v][ni * 16 + rl] = acc[mi][ni][v];
  __syncthreads();
  for (int i = tid; i < 1024; i += 256) {
    float s = 0.0f;
#pragma unroll
    for (int w = 0; w < 8; ++w) s += part[w][i >> 5][i & 31];
    S[(size_t)b * 1024 + i] = s;
  }
}

// ---------------------------------------------------------------------------
// small support kernels
// ---------------------------------------------------------------------------
__global__ __launch_bounds__(256)
void k_colnorm(const float* __restrict__ X, float* __restrict__ nrm, int ntok)
{
  __shared__ float red[256];
  const int c = blockIdx.x;
  const int b = blockIdx.y;
  const float* xp = X + (size_t)b * ntok * WC;
  float s = 0.0f;
  for (int t = threadIdx.x; t < ntok; t += 256) {
    float v = xp[(size_t)t * WC + c];
    s += v * v;
  }
  red[threadIdx.x] = s;
  __syncthreads();
  for (int k = 128; k > 0; k >>= 1) {
    if (threadIdx.x < k) red[threadIdx.x] += red[threadIdx.x + k];
    __syncthreads();
  }
  if (threadIdx.x == 0) nrm[b * 32 + c] = fmaxf(sqrtf(red[0]), 1e-12f);
}

__global__ void k_softmax31(const float* __restrict__ S, const float* __restrict__ nk,
                            const float* __restrict__ nq, const float* __restrict__ rescale,
                            float* __restrict__ P)
{
  const int b = blockIdx.x;
  const int i = threadIdx.x;
  if (i >= WC) return;
  const float rs = rescale[0];
  float row[WC];
  float mx = -1e30f;
  for (int j = 0; j < WC; ++j) {
    float v = S[(size_t)b * 1024 + i * 32 + j] / (nk[b * 32 + i] * nq[b * 32 + j]) * rs;
    row[j] = v;
    mx = fmaxf(mx, v);
  }
  float sum = 0.0f;
  for (int j = 0; j < WC; ++j) { row[j] = __expf(row[j] - mx); sum += row[j]; }
  const float inv = 1.0f / sum;
  for (int j = 0; j < WC; ++j) P[(size_t)b * 1024 + i * 32 + j] = row[j] * inv;
}

// f32 [nt,C] -> zero-padded f16 [nt,ldp]
__global__ void k_cvt_f16pad(const float* __restrict__ in, _Float16* __restrict__ out,
                             int nt, int C, int ldp)
{
  int idx = blockIdx.x * blockDim.x + threadIdx.x;
  if (idx >= nt * ldp) return;
  int c = idx % ldp, t = idx / ldp;
  float v = (c < C) ? in[(size_t)t * C + c] : 0.0f;
  out[(size_t)t * ldp + c] = (_Float16)v;
}

// strided product -> zero-padded f16 [nt,ldp]
__global__ void k_mul_f16pad(const float* __restrict__ a, int as, int ac,
                             const float* __restrict__ b, int bs, int bc,
                             _Float16* __restrict__ o, int ldp, int nt, int C)
{
  int idx = blockIdx.x * blockDim.x + threadIdx.x;
  if (idx >= nt * ldp) return;
  int c = idx % ldp, t = idx / ldp;
  float v = (c < C) ? a[(size_t)t * as + ac + c] * b[(size_t)t * bs + bc + c] : 0.0f;
  o[(size_t)t * ldp + c] = (_Float16)v;
}

__global__ void k_add3(const float* __restrict__ a, const float* __restrict__ b,
                       const float* __restrict__ c, float* __restrict__ o, int n)
{
  int i = blockIdx.x * blockDim.x + threadIdx.x;
  if (i < n) o[i] = a[i] + b[i] + c[i];
}

// depthwise 3x3 on token-major [t, C] with fused activation
// act: 0 = none, 1 = gelu, 2 = tanh(conv)+input (eMLP residual branch)
__global__ void k_dwconv3x3(const float* __restrict__ in, int inS, int inC0,
                            const float* __restrict__ w,
                            float* __restrict__ out, int outS, int outC0,
                            int Cn, int act)
{
  int idx = blockIdx.x * blockDim.x + threadIdx.x;
  if (idx >= NTOK * Cn) return;
  int c = idx % Cn;
  int t = idx / Cn;
  int px = t % WW, py = (t / WW) % WH, b = t / (WW * WH);
  float s = 0.0f;
#pragma unroll
  for (int dy = 0; dy < 3; ++dy) {
    int yy = py + dy - 1;
    if (yy < 0 || yy >= WH) continue;
#pragma unroll
    for (int dx = 0; dx < 3; ++dx) {
      int xx = px + dx - 1;
      if (xx < 0 || xx >= WW) continue;
      s += in[((size_t)(b * WH + yy) * WW + xx) * inS + inC0 + c] * w[c * 9 + dy * 3 + dx];
    }
  }
  if (act == 1) s = gelu_exact(s);
  else if (act == 2) s = tanhf(s) + in[((size_t)(b * WH + py) * WW + px) * inS + inC0 + c];
  out[((size_t)(b * WH + py) * WW + px) * outS + outC0 + c] = s;
}

// layernorm over 31 channels -> zero-padded f16 [t, LDQ]
__global__ void k_layernorm_f16(const float* __restrict__ x, const float* __restrict__ g,
                                const float* __restrict__ bta, _Float16* __restrict__ o,
                                int nt)
{
  int t = blockIdx.x * blockDim.x + threadIdx.x;
  if (t >= nt) return;
  const float* xp = x + (size_t)t * WC;
  float m = 0.0f;
  for (int c = 0; c < WC; ++c) m += xp[c];
  m /= (float)WC;
  float v = 0.0f;
  for (int c = 0; c < WC; ++c) { float d = xp[c] - m; v += d * d; }
  v /= (float)WC;
  float inv = rsqrtf(v + 1e-5f);
  for (int c = 0; c < WC; ++c)
    o[(size_t)t * LDQ + c] = (_Float16)((xp[c] - m) * inv * g[c] + bta[c]);
  o[(size_t)t * LDQ + WC] = (_Float16)0.0f;
}

// embed: x NCHW(3) --3x3 pad1--> token-major [t, 31]
__global__ void k_embed_conv3x3(const float* __restrict__ x, const float* __restrict__ w,
                                float* __restrict__ out)
{
  int idx = blockIdx.x * blockDim.x + threadIdx.x;
  if (idx >= NTOK * WC) return;
  int co = idx % WC;
  int t = idx / WC;
  int px = t % WW, py = (t / WW) % WH, b = t / (WW * WH);
  float s = 0.0f;
  for (int ci = 0; ci < 3; ++ci)
#pragma unroll
    for (int dy = 0; dy < 3; ++dy) {
      int yy = py + dy - 1;
      if (yy < 0 || yy >= WH) continue;
#pragma unroll
      for (int dx = 0; dx < 3; ++dx) {
        int xx = px + dx - 1;
        if (xx < 0 || xx >= WW) continue;
        s += x[(((size_t)b * 3 + ci) * WH + yy) * WW + xx] * w[((co * 3 + ci) * 3 + dy) * 3 + dx];
      }
    }
  out[(size_t)t * WC + co] = s;
}

// NCHW(31) -> token-major [t, 31]
__global__ void k_nchw2tok(const float* __restrict__ in, float* __restrict__ out)
{
  int idx = blockIdx.x * blockDim.x + threadIdx.x;
  if (idx >= NTOK * WC) return;
  int c = idx % WC;
  int t = idx / WC;
  int px = t % WW, py = (t / WW) % WH, b = t / (WW * WH);
  out[(size_t)t * WC + c] = in[(((size_t)b * WC + c) * WH + py) * WW + px];
}

// map: token-major [t,31] --3x3 pad1--> NCHW(3), + input residual
__global__ void k_map_conv3x3_res(const float* __restrict__ fea, const float* __restrict__ w,
                                  const float* __restrict__ xin, float* __restrict__ out)
{
  int idx = blockIdx.x * blockDim.x + threadIdx.x;
  if (idx >= WB * 3 * WH * WW) return;
  int px = idx % WW;
  int py = (idx / WW) % WH;
  int co = (idx / (WW * WH)) % 3;
  int b  = idx / (WW * WH * 3);
  float s = 0.0f;
  for (int ci = 0; ci < WC; ++ci)
#pragma unroll
    for (int dy = 0; dy < 3; ++dy) {
      int yy = py + dy - 1;
      if (yy < 0 || yy >= WH) continue;
#pragma unroll
      for (int dx = 0; dx < 3; ++dx) {
        int xx = px + dx - 1;
        if (xx < 0 || xx >= WW) continue;
        s += fea[(((size_t)b * WH + yy) * WW + xx) * WC + ci] * w[((co * WC + ci) * 3 + dy) * 3 + dx];
      }
    }
  out[idx] = s + xin[idx];
}

// ---------------------------------------------------------------------------
// host-side orchestration
// ---------------------------------------------------------------------------
extern "C" void kernel_launch(void* const* d_in, const int* in_sizes, int n_in,
                              void* d_out, int out_size, void* d_ws, size_t ws_size,
                              hipStream_t stream)
{
  (void)in_sizes; (void)out_size; (void)ws_size;
  auto in = [&](int i) -> const float* {
    if (i < 0) i = 0;
    if (i >= n_in) i = n_in - 1;
    return (const float*)d_in[i];
  };
  const float* x    = in(0);
  const float* illu = in(1);

  // JAX pytree (sorted-key) flatten order, leaves start at index 2:
  // bottleneck(60) | dec(96) | embed(1) | enc(114) | map(1)
  const int IDX_EMBED = 158;
  const int IDX_MAP   = 273;
  const int IGAB0     = 168;  // enc[0].igab, 2 blocks x 15 leaves
  enum { L_DW = 0, L_DW1, L_DW2, L_PIN, L_POUT, L_LNB, L_LNG,
         L_POS1, L_POS2, L_PROJB, L_PROJW, L_RESCALE, L_WK, L_WQ, L_WV };

  // workspace carve-up
  char* wsp = (char*)d_ws;
  size_t off = 0;
  auto alloc = [&](size_t bytes) -> void* {
    void* p = (void*)(wsp + off);
    off += (bytes + 255) & ~(size_t)255;
    return p;
  };
  float* fea   = (float*)alloc((size_t)NTOK * WC * 4);
  float* illuT = (float*)alloc((size_t)NTOK * WC * 4);
  float* bq    = (float*)alloc((size_t)NTOK * WC * 4);
  float* bk    = (float*)alloc((size_t)NTOK * WC * 4);
  float* bv    = (float*)alloc((size_t)NTOK * WC * 4);
  float* bt    = (float*)alloc((size_t)NTOK * WC * 4);
  float* bo    = (float*)alloc((size_t)NTOK * WC * 4);
  float* h1    = (float*)alloc((size_t)NTOK * ECH * 4);
  float* h2    = (float*)alloc((size_t)NTOK * ECH * 4);
  _Float16* feah = (_Float16*)alloc((size_t)NTOK * LDQ * 2);
  _Float16* bth  = (_Float16*)alloc((size_t)NTOK * LDQ * 2);
  _Float16* lnh  = (_Float16*)alloc((size_t)NTOK * LDQ * 2);
  _Float16* xmh  = (_Float16*)alloc((size_t)NTOK * LDH * 2);
  float* S     = (float*)alloc((size_t)WB * 1024 * 4);
  float* P     = (float*)alloc((size_t)WB * 1024 * 4);
  float* nq    = (float*)alloc((size_t)WB * 32 * 4);
  float* nk    = (float*)alloc((size_t)WB * 32 * 4);

  const int EW = NTOK * WC;
  auto ew_grid = [&](int n) { return dim3((unsigned)((n + 255) / 256)); };

  auto gemm16 = [&](const _Float16* A16, int lda, const float* W, const float* bias,
                    const float* resid, float* C, int M, int N, int K, int ldw, int wT) {
    dim3 g((unsigned)((M + TM - 1) / TM), (unsigned)((N + TN - 1) / TN));
    k_gemm_wmma_tdm<<<g, 128, 0, stream>>>(A16, lda, W, bias, resid, C, M, N, K, ldw, wT);
  };
  auto gemm32 = [&](const float* A, const float* W, const float* bias, const float* resid,
                    float* C, int M, int N, int K, int ldw, int wT) {
    dim3 g((unsigned)((M + TM - 1) / TM), (unsigned)((N + TN - 1) / TN));
    k_gemm_wmma<<<g, 128, 0, stream>>>(A, W, bias, resid, C, M, N, K, ldw, wT);
  };

  k_embed_conv3x3<<<ew_grid(EW), 256, 0, stream>>>(x, in(IDX_EMBED), fea);
  k_nchw2tok<<<ew_grid(EW), 256, 0, stream>>>(illu, illuT);

  for (int blk = 0; blk < 2; ++blk) {
    const int base = IGAB0 + blk * 15;

    // ---------------- MSA ----------------
    // fea -> padded f16 once, feeds all three projection GEMMs via TDM
    k_cvt_f16pad<<<ew_grid(NTOK * LDQ), 256, 0, stream>>>(fea, feah, NTOK, WC, LDQ);
    gemm16(feah, LDQ, in(base + L_WQ), nullptr, nullptr, bq, NTOK, WC, WC, WC, 0);
    gemm16(feah, LDQ, in(base + L_WK), nullptr, nullptr, bk, NTOK, WC, WC, WC, 0);
    gemm16(feah, LDQ, in(base + L_WV), nullptr, nullptr, bv, NTOK, WC, WC, WC, 0);
    k_colnorm<<<dim3(WC, WB), 256, 0, stream>>>(bq, nq, NPB);
    k_colnorm<<<dim3(WC, WB), 256, 0, stream>>>(bk, nk, NPB);
    k_gram_wmma<<<WB, 256, 0, stream>>>(bk, bq, S, NPB);
    k_softmax31<<<WB, 32, 0, stream>>>(S, nk, nq, in(base + L_RESCALE), P);
    // v = v_inp * illu (written f16-padded for the TDM GEMM)
    k_mul_f16pad<<<ew_grid(NTOK * LDQ), 256, 0, stream>>>(bv, WC, 0, illuT, WC, 0,
                                                          bth, LDQ, NTOK, WC);
    // o[t,i] = sum_j v[t,j] * P[i,j]  (P as transposed 31x31 weight, per batch)
    for (int b = 0; b < WB; ++b)
      gemm16(bth + (size_t)b * NPB * LDQ, LDQ, P + (size_t)b * 1024, nullptr, nullptr,
             bo + (size_t)b * NPB * WC, NPB, WC, WC, 32, 1);
    gemm32(bo, in(base + L_PROJW), in(base + L_PROJB), nullptr, bq, NTOK, WC, WC, WC, 0);
    // positional branch: dw3x3 -> gelu -> dw3x3 on v_inp
    k_dwconv3x3<<<ew_grid(EW), 256, 0, stream>>>(bv, WC, 0, in(base + L_POS1), bk, WC, 0, WC, 1);
    k_dwconv3x3<<<ew_grid(EW), 256, 0, stream>>>(bk, WC, 0, in(base + L_POS2), bt, WC, 0, WC, 0);
    // fea = fea + out_c + pos_emb
    k_add3<<<ew_grid(EW), 256, 0, stream>>>(fea, bq, bt, fea, EW);

    // ---------------- eMLP ----------------
    k_layernorm_f16<<<ew_grid(NTOK), 256, 0, stream>>>(fea, in(base + L_LNG),
                                                       in(base + L_LNB), lnh, NTOK);
    gemm16(lnh, LDQ, in(base + L_PIN), nullptr, nullptr, h1, NTOK, ECH, WC, WC, 1);
    k_dwconv3x3<<<ew_grid(NTOK * ECH), 256, 0, stream>>>(h1, ECH, 0, in(base + L_DW), h2, ECH, 0, ECH, 0);
    // x1 = tanh(dw1(x1)) + x1 ; x2 = tanh(dw2(x2)) + x2   (channel slices of h2)
    k_dwconv3x3<<<ew_grid(NTOK * EHID), 256, 0, stream>>>(h2, ECH, 0,    in(base + L_DW1), h1, ECH, 0,    EHID, 2);
    k_dwconv3x3<<<ew_grid(NTOK * EHID), 256, 0, stream>>>(h2, ECH, EHID, in(base + L_DW2), h1, ECH, EHID, EHID, 2);
    // x1*x2 -> f16-padded [t, 84] for the TDM GEMM
    k_mul_f16pad<<<ew_grid(NTOK * LDH), 256, 0, stream>>>(h1, ECH, 0, h1, ECH, EHID,
                                                          xmh, LDH, NTOK, EHID);
    // fea += pout(x1*x2)
    gemm16(xmh, LDH, in(base + L_POUT), nullptr, fea, fea, NTOK, WC, EHID, EHID, 1);
  }

  k_map_conv3x3_res<<<ew_grid(WB * 3 * WH * WW), 256, 0, stream>>>(fea, in(IDX_MAP), x, (float*)d_out);
}
